// MultiHeadedAttention_17540646436925
// MI455X (gfx1250) — compile-verified, matrix operand order checked by reading, not run on hardware
//
#include <hip/hip_runtime.h>
#include <math.h>

// ---------------------------------------------------------------------------
// MI455X (gfx1250) multi-head "channel attention" pipeline, bf16 WMMA.
//
// Roofline: ~103 GFLOP, ~270 MB HBM traffic -> AI ~380 FLOP/B, near the
// bf16-WMMA / 23.3 TB/s balance point. All matmuls use
// v_wmma_f32_16x16x32_bf16; global->LDS staging uses CDNA5
// global_load_async_to_lds_b128 (ASYNCcnt) double-buffered so the next
// tile's fetch overlaps the current tile's WMMAs.
// ---------------------------------------------------------------------------

typedef __bf16 bf16;
typedef __attribute__((ext_vector_type(16))) __bf16        v16bf;
typedef __attribute__((ext_vector_type(8)))  float         v8f;
typedef __attribute__((ext_vector_type(4)))  unsigned int  u32x4;
typedef __attribute__((ext_vector_type(4)))  float         f32x4;

#define LDS_STRIDE 40   // 32 bf16 + 8 pad: rows stay 16B-aligned, banks spread

// CDNA5 async global->LDS copy, 16 bytes per lane. VDST = VGPR holding the
// wave-relative LDS byte offset (low 32 bits of a flat shared pointer, per
// ISA aperture rules); VADDR = 64-bit global address; tracked by ASYNCcnt.
__device__ __forceinline__ void async_copy16(const void* gsrc, void* lds_dst) {
  unsigned lofs = (unsigned)(unsigned long long)(uintptr_t)lds_dst;
  asm volatile("global_load_async_to_lds_b128 %0, %1, off"
               :: "v"(lofs), "v"(gsrc)
               : "memory");
}
__device__ __forceinline__ void wait_async0() {
  asm volatile("s_wait_asynccnt 0x0" ::: "memory");
}

// Load one 16x32 bf16 WMMA fragment (A-layout; B uses the symmetric N-major
// layout, so the same loader works on an N-major LDS tile).
// Per CDNA5 ISA 7.12.2: lanes 0-15 row r hold K=0..7 (v0-3) and K=16..23
// (v4-7); lanes 16-31 hold K=8..15 and K=24..31.
__device__ __forceinline__ v16bf frag_from_lds(const bf16* base, int lane) {
  const int r    = lane & 15;
  const int half = lane >> 4;
  const bf16* p  = base + r * LDS_STRIDE + half * 8;
  v16bf f;
  u32x4* d = reinterpret_cast<u32x4*>(&f);
  d[0] = *reinterpret_cast<const u32x4*>(p);       // K chunk 0 (16B)
  d[1] = *reinterpret_cast<const u32x4*>(p + 16);  // K chunk 1 (16B)
  return f;
}

__device__ __forceinline__ void storeC(float* p, float v) { *p = v; }
__device__ __forceinline__ void storeC(bf16*  p, float v) { *p = (bf16)v; }

// ---------------------------------------------------------------------------
// Generic bf16 GEMM:  C[z] = alpha * A[z](MxK, row-major) * B[z](KxN)
//   B_IS_NK = true  : B stored N-major -> async-copied straight into LDS
//   B_IS_NK = false : B stored K-major -> reg-staged + transpose scatter
// z decomposes as (h = z>>4, b = z&15); per-operand strides sXh/sXb (elems).
// Block tile 128x128, BK=32, 8 waves; wave tile 64x32 = 4x2 WMMA tiles.
// Double-buffered LDS, one barrier per k-iteration; next tile in flight
// (ASYNCcnt / staging regs) while the current tile feeds the WMMAs.
// All dims divide the tile sizes exactly -> no bounds checks.
// ---------------------------------------------------------------------------
template <typename CT, bool B_IS_NK>
__global__ void __launch_bounds__(256)
gemm_bf16_wmma(const bf16* __restrict__ A, const bf16* __restrict__ B,
               CT* __restrict__ C,
               int lda, int ldb, int ldc, int K,
               long sAh, long sAb, long sBh, long sBb,
               long sCh, long sCb, float alpha) {
  __shared__ bf16 Alds[2][128 * LDS_STRIDE];  // M x K tile, row-major
  __shared__ bf16 Blds[2][128 * LDS_STRIDE];  // N x K tile, N-major

  const int t    = threadIdx.x;
  const int lane = t & 31;
  const int wave = t >> 5;
  const int wm0  = (wave >> 2) * 64;   // wave row offset in block tile
  const int wn0  = (wave & 3) * 32;    // wave col offset in block tile
  const int m0   = blockIdx.y * 128;
  const int n0   = blockIdx.x * 128;
  const int z    = blockIdx.z;
  const int hh   = z >> 4;
  const int bb   = z & 15;

  const bf16* Az = A + hh * sAh + bb * sAb;
  const bf16* Bz = B + hh * sBh + bb * sBb;
  CT*         Cz = C + hh * sCh + bb * sCb;

  const int kIters = K >> 5;

  // ---- per-thread chunk mapping for 128x32 tiles (512 16B chunks) ---------
  const int arow0 = t >> 2;          // rows 0..63 (chunk 1 adds +64)
  const int acol  = (t & 3) * 8;     // k offset 0/8/16/24
  const bf16* aG0 = Az + (long)(m0 + arow0) * lda + acol;
  const bf16* aG1 = aG0 + (long)64 * lda;

  // N-major B: identical mapping against ldb
  const bf16* nG0 = nullptr;
  const bf16* nG1 = nullptr;
  // K-major B (transpose path): 16B along N, scattered into LDS
  const int   krow0 = t >> 4;        // 0..15 (chunk 1 adds +16)
  const int   nch   = (t & 15) * 8;  // 0..120
  const bf16* tG0 = nullptr;
  const bf16* tG1 = nullptr;
  u32x4 brA = {}, brB = {};

  if constexpr (B_IS_NK) {
    nG0 = Bz + (long)(n0 + arow0) * ldb + acol;
    nG1 = nG0 + (long)64 * ldb;
  } else {
    tG0 = Bz + (long)krow0 * ldb + n0 + nch;
    tG1 = tG0 + (long)16 * ldb;
    brA = *reinterpret_cast<const u32x4*>(tG0);   // stage tile 0
    brB = *reinterpret_cast<const u32x4*>(tG1);
  }

  // ---- async prologue: tile 0 -> buffer 0 ---------------------------------
  async_copy16(aG0, &Alds[0][arow0 * LDS_STRIDE + acol]);
  async_copy16(aG1, &Alds[0][(arow0 + 64) * LDS_STRIDE + acol]);
  if constexpr (B_IS_NK) {
    async_copy16(nG0, &Blds[0][arow0 * LDS_STRIDE + acol]);
    async_copy16(nG1, &Blds[0][(arow0 + 64) * LDS_STRIDE + acol]);
  }

  v8f acc[4][2];
#pragma unroll
  for (int mt = 0; mt < 4; ++mt)
#pragma unroll
    for (int nt = 0; nt < 2; ++nt)
#pragma unroll
      for (int r = 0; r < 8; ++r) acc[mt][nt][r] = 0.0f;

  for (int it = 0; it < kIters; ++it) {
    const int buf = it & 1;

    if constexpr (!B_IS_NK) {
      // Scatter staged tile `it` into Blds[buf] (N-major), then stage it+1
      // so those global loads overlap this iteration's WMMAs.
      const bf16* s0 = reinterpret_cast<const bf16*>(&brA);
      const bf16* s1 = reinterpret_cast<const bf16*>(&brB);
#pragma unroll
      for (int j = 0; j < 8; ++j) {
        Blds[buf][(nch + j) * LDS_STRIDE + krow0]      = s0[j];
        Blds[buf][(nch + j) * LDS_STRIDE + krow0 + 16] = s1[j];
      }
      if (it + 1 < kIters) {
        tG0 += (long)32 * ldb;
        tG1 += (long)32 * ldb;
        brA = *reinterpret_cast<const u32x4*>(tG0);
        brB = *reinterpret_cast<const u32x4*>(tG1);
      }
    }

    wait_async0();      // tile `it` async copies have landed in LDS
    __syncthreads();    // (also flushes this wave's scatter via DScnt wait)

    if (it + 1 < kIters) {
      const int nb = buf ^ 1;
      aG0 += 32;
      aG1 += 32;
      async_copy16(aG0, &Alds[nb][arow0 * LDS_STRIDE + acol]);
      async_copy16(aG1, &Alds[nb][(arow0 + 64) * LDS_STRIDE + acol]);
      if constexpr (B_IS_NK) {
        nG0 += 32;
        nG1 += 32;
        async_copy16(nG0, &Blds[nb][arow0 * LDS_STRIDE + acol]);
        async_copy16(nG1, &Blds[nb][(arow0 + 64) * LDS_STRIDE + acol]);
      }
    }

    // ---- WMMA pass: 4x2 tiles, one 16x16x32 per tile ----------------------
    v16bf af[4];
#pragma unroll
    for (int mt = 0; mt < 4; ++mt)
      af[mt] = frag_from_lds(&Alds[buf][(wm0 + mt * 16) * LDS_STRIDE], lane);
    v16bf bfr[2];
#pragma unroll
    for (int nt = 0; nt < 2; ++nt)
      bfr[nt] = frag_from_lds(&Blds[buf][(wn0 + nt * 16) * LDS_STRIDE], lane);

#pragma unroll
    for (int mt = 0; mt < 4; ++mt)
#pragma unroll
      for (int nt = 0; nt < 2; ++nt)
        acc[mt][nt] = __builtin_amdgcn_wmma_f32_16x16x32_bf16(
            false, af[mt], false, bfr[nt], (short)0, acc[mt][nt], false, false);
  }

  // ---- store: C layout per ISA: lane = (halfM, col); VGPR r -> row ---------
  const int colL = lane & 15;
  const int rowH = (lane >> 4) * 8;
#pragma unroll
  for (int mt = 0; mt < 4; ++mt)
#pragma unroll
    for (int nt = 0; nt < 2; ++nt) {
      CT* cp = Cz + (long)(m0 + wm0 + mt * 16 + rowH) * ldc + (n0 + wn0 + nt * 16 + colL);
#pragma unroll
      for (int r = 0; r < 8; ++r)
        storeC(cp + (long)r * ldc, acc[mt][nt][r] * alpha);
    }
}

// ---------------------------------------------------------------------------
// Elementwise helpers
// ---------------------------------------------------------------------------
__global__ void __launch_bounds__(256)
cvt_f32_to_bf16(const float* __restrict__ in, bf16* __restrict__ out, int n) {
  const int i = (blockIdx.x * 256 + threadIdx.x) * 4;
  if (i >= n) return;
  f32x4 v = *reinterpret_cast<const f32x4*>(in + i);
#pragma unroll
  for (int j = 0; j < 4; ++j) out[i + j] = (bf16)v[j];
}

__global__ void __launch_bounds__(256)
sub_to_bf16(const float* __restrict__ x, const float* __restrict__ o,
            bf16* __restrict__ u, int n) {
  const int i = (blockIdx.x * 256 + threadIdx.x) * 4;
  if (i >= n) return;
  f32x4 a = *reinterpret_cast<const f32x4*>(x + i);
  f32x4 b = *reinterpret_cast<const f32x4*>(o + i);
#pragma unroll
  for (int j = 0; j < 4; ++j) u[i + j] = (bf16)(a[j] - b[j]);
}

// One wave32 per row of 512; lane handles 16 strided elements; shuffle reduce.
__global__ void __launch_bounds__(256)
softmax_rows512(const float* __restrict__ S, bf16* __restrict__ P, float scale) {
  const long row  = (long)blockIdx.x * 8 + (threadIdx.x >> 5);
  const int  lane = threadIdx.x & 31;
  const float* s = S + row * 512;
  bf16*        p = P + row * 512;

  float x[16];
  float m = -3.4e38f;
#pragma unroll
  for (int j = 0; j < 16; ++j) {
    x[j] = s[lane + j * 32] * scale;
    m = fmaxf(m, x[j]);
  }
#pragma unroll
  for (int off = 16; off > 0; off >>= 1) m = fmaxf(m, __shfl_xor(m, off, 32));
  float sum = 0.0f;
#pragma unroll
  for (int j = 0; j < 16; ++j) {
    x[j] = __expf(x[j] - m);
    sum += x[j];
  }
#pragma unroll
  for (int off = 16; off > 0; off >>= 1) sum += __shfl_xor(sum, off, 32);
  const float inv = 1.0f / sum;
#pragma unroll
  for (int j = 0; j < 16; ++j) p[lane + j * 32] = (bf16)(x[j] * inv);
}

// ---------------------------------------------------------------------------
// Orchestration. Shapes fixed by the reference: B=16, D=512, N=2048, H=8.
// Workspace (~390 MB): bf16 weights + bf16 X copies + bf16 Q/K/V + f32 scores
// (region reused for O) + bf16 P; U reuses the Xq bf16 buffer.
// ---------------------------------------------------------------------------
extern "C" void kernel_launch(void* const* d_in, const int* in_sizes, int n_in,
                              void* d_out, int out_size, void* d_ws, size_t ws_size,
                              hipStream_t stream) {
  (void)in_sizes; (void)n_in; (void)out_size; (void)ws_size;

  const float* Xq = (const float*)d_in[0];
  const float* Xk = (const float*)d_in[1];
  const float* Xv = (const float*)d_in[2];
  const float* Wq = (const float*)d_in[3];
  const float* Wk = (const float*)d_in[4];
  const float* Wv = (const float*)d_in[5];
  const float* Wo = (const float*)d_in[6];
  float* out = (float*)d_out;

  const int B = 16, D = 512, N = 2048, H = 8, Nh = 256;
  const long XN = (long)B * D * N;      // 16,777,216
  const long WN = (long)D * D;          // 262,144
  const long SB = (long)D * N;          // per-batch stride in X/Q/K/V/out
  const long SN = (long)H * B * D * D;  // 33,554,432 score elements

  char* ws = (char*)d_ws;
  size_t off = 0;
  auto take = [&](size_t bytes) -> char* {
    char* p = ws + off;
    off += (bytes + 255) & ~(size_t)255;
    return p;
  };

  bf16* Wqb = (bf16*)take(WN * 2);
  bf16* Wkb = (bf16*)take(WN * 2);
  bf16* Wvb = (bf16*)take(WN * 2);
  bf16* Wob = (bf16*)take(WN * 2);
  bf16* Xqb = (bf16*)take(XN * 2);
  bf16* Xkb = (bf16*)take(XN * 2);
  bf16* Xvb = (bf16*)take(XN * 2);
  bf16* Qb  = (bf16*)take(XN * 2);
  bf16* Kb  = (bf16*)take(XN * 2);
  bf16* Vb  = (bf16*)take(XN * 2);
  float* S  = (float*)take(SN * 4);     // scores f32; first half reused for O
  bf16*  P  = (bf16*)take(SN * 2);      // softmax probs bf16
  float* O  = S;                        // attn output f32 (64 MiB into 128 MiB)
  bf16*  U  = Xqb;                      // (XQ - attn) bf16, reuses Xq bf16

  const dim3 blk(256);

  // 1) fp32 -> bf16 conversions
  cvt_f32_to_bf16<<<(int)(WN / 1024), blk, 0, stream>>>(Wq, Wqb, (int)WN);
  cvt_f32_to_bf16<<<(int)(WN / 1024), blk, 0, stream>>>(Wk, Wkb, (int)WN);
  cvt_f32_to_bf16<<<(int)(WN / 1024), blk, 0, stream>>>(Wv, Wvb, (int)WN);
  cvt_f32_to_bf16<<<(int)(WN / 1024), blk, 0, stream>>>(Wo, Wob, (int)WN);
  cvt_f32_to_bf16<<<(int)(XN / 1024), blk, 0, stream>>>(Xq, Xqb, (int)XN);
  cvt_f32_to_bf16<<<(int)(XN / 1024), blk, 0, stream>>>(Xk, Xkb, (int)XN);
  cvt_f32_to_bf16<<<(int)(XN / 1024), blk, 0, stream>>>(Xv, Xvb, (int)XN);

  // 2) Projections: Q/K/V[b] = W * X[b]   (M=512, K=512, N=2048, z=b)
  const dim3 gProj(N / 128, D / 128, B);
  gemm_bf16_wmma<bf16, false><<<gProj, blk, 0, stream>>>(
      Wqb, Xqb, Qb, D, N, N, D, 0, 0, 0, SB, 0, SB, 1.0f);
  gemm_bf16_wmma<bf16, false><<<gProj, blk, 0, stream>>>(
      Wkb, Xkb, Kb, D, N, N, D, 0, 0, 0, SB, 0, SB, 1.0f);
  gemm_bf16_wmma<bf16, false><<<gProj, blk, 0, stream>>>(
      Wvb, Xvb, Vb, D, N, N, D, 0, 0, 0, SB, 0, SB, 1.0f);

  // 3) Scores: S[h,b] = Q_hb * K_hb^T  (M=512, K=256, N=512; K^T is N-major)
  const dim3 gScore(D / 128, D / 128, H * B);
  gemm_bf16_wmma<float, true><<<gScore, blk, 0, stream>>>(
      Qb, Kb, S, N, N, D, Nh,
      (long)Nh, SB, (long)Nh, SB, (long)B * D * D, (long)D * D, 1.0f);

  // 4) Softmax rows (scale 1/sqrt(Nh) = 1/16), f32 -> bf16 probs
  softmax_rows512<<<(int)(SN / (512 * 8)), blk, 0, stream>>>(S, P, 1.0f / 16.0f);

  // 5) O[h,b] = P_hb * V_hb  (M=512, K=512, N=256) written into (B,D,N) layout
  const dim3 gOut(Nh / 128, D / 128, H * B);
  gemm_bf16_wmma<float, false><<<gOut, blk, 0, stream>>>(
      P, Vb, O, D, N, N, D,
      (long)B * D * D, (long)D * D, (long)Nh, SB, (long)Nh, SB, 1.0f);

  // 6) U = bf16(XQ - O)
  sub_to_bf16<<<(int)(XN / 1024), blk, 0, stream>>>(Xq, O, U, (int)XN);

  // 7) out[b] = Wo * U[b]  (M=512, K=512, N=2048, f32 output)
  gemm_bf16_wmma<float, false><<<gProj, blk, 0, stream>>>(
      Wob, U, out, D, N, N, D, 0, 0, 0, SB, 0, SB, 1.0f);
}
